// DualAxisAggAttn_86174223827340
// MI455X (gfx1250) — compile-verified, matrix operand order checked by reading, not run on hardware
//
#include <hip/hip_runtime.h>

// Problem constants
#define BB 16
#define CC 256
#define HH 80
#define WW 80
#define LL 80                 // length along the attention axis
#define PLANE 6400            // H*W
#define XROW_STRIDE 81        // fp32 row-major pad (anti bank-conflict)
#define XT_STRIDE  264        // bf16 transposed pad (anti bank-conflict)
#define BN_EPS 1e-5f

typedef __attribute__((ext_vector_type(16))) __bf16 v16bf;
typedef __attribute__((ext_vector_type(8)))  __bf16 v8bf;
typedef __attribute__((ext_vector_type(8)))  float  v8f;

// ---------------------------------------------------------------------------
// Repack Wv (qkv rows 1+C .. 2C) into per-lane WMMA A-operand order (bf16).
// A tile = 16x32 (MxK). Lane L (m=L%16, kh=L/16) owns 16 bf16:
//   j<8  -> K = kt*32 + kh*8 + j          (VGPR0..3)
//   j>=8 -> K = kt*32 + 16 + kh*8 + (j-8) (VGPR4..7)
// dst index = ((mt*8 + kt)*32 + L)*16 + j  == linear r, for both axes.
// ---------------------------------------------------------------------------
__global__ __launch_bounds__(256) void prep_wv_kernel(
    const float* __restrict__ wW, const float* __restrict__ wH,
    __bf16* __restrict__ dst)
{
    int gid = blockIdx.x * 256 + threadIdx.x;
    if (gid >= 2 * CC * CC) return;
    int axis = gid / (CC * CC);
    int r    = gid % (CC * CC);
    int j  = r & 15;
    int L  = (r >> 4) & 31;
    int kt = (r >> 9) & 7;
    int mt = r >> 12;
    int m  = mt * 16 + (L & 15);
    int kh = L >> 4;
    int klocal = (j < 8) ? (kh * 8 + j) : (16 + kh * 8 + (j - 8));
    int k = kt * 32 + klocal;
    const float* wsrc = axis ? wH : wW;
    dst[gid] = (__bf16)wsrc[(size_t)(1 + CC + m) * CC + k];
}

// ---------------------------------------------------------------------------
// One (b,row) slice: q -> softmax -> xs -> ctx (VALU fp32),
// v = Wv @ x_slice via v_wmma_f32_16x16x32_bf16, fused epilogue
// out = x + sigmoid(v) * ctx.
// axis=W: rowMul=W, istride=1 ; axis=H: rowMul=1, istride=W.
// ---------------------------------------------------------------------------
__global__ __launch_bounds__(256) void attn_axis_kernel(
    const float* __restrict__ src, float* __restrict__ dst,
    const float* __restrict__ qkv_w, const float* __restrict__ qkv_b,
    const __bf16* __restrict__ wv_tiled,
    int rowMul, int istride)
{
    __shared__ float  xrow[CC * XROW_STRIDE];  // fp32, [c][i]
    __shared__ __bf16 xT[LL * XT_STRIDE];      // bf16, [i][c] for WMMA B
    __shared__ float  scores[LL];
    __shared__ float  xsum[CC];
    __shared__ float  ctxv[CC];

    const int tid = threadIdx.x;
    const int b   = blockIdx.x / LL;
    const int row = blockIdx.x % LL;
    const size_t base = (size_t)b * CC * PLANE + (size_t)row * rowMul;

    // Stage x slice into LDS (fp32 + transposed bf16 copy)
    for (int idx = tid; idx < CC * LL; idx += 256) {
        int c = idx / LL, i = idx % LL;
        float v = src[base + (size_t)c * PLANE + (size_t)i * istride];
        xrow[c * XROW_STRIDE + i] = v;
        xT[i * XT_STRIDE + c]     = (__bf16)v;
    }
    __syncthreads();

    // q[i] = Wq . x[:,i] + bq   (Wq = row 0)
    if (tid < LL) {
        float acc = qkv_b[0];
        #pragma unroll 4
        for (int c = 0; c < CC; ++c)
            acc += qkv_w[c] * xrow[c * XROW_STRIDE + tid];
        scores[tid] = acc;
    }
    __syncthreads();

    // softmax over the 80 positions (redundant broadcast reductions: cheap)
    float qmax = -1e30f;
    for (int i = 0; i < LL; ++i) qmax = fmaxf(qmax, scores[i]);
    float e = 0.f;
    if (tid < LL) e = __expf(scores[tid] - qmax);
    __syncthreads();
    if (tid < LL) scores[tid] = e;
    __syncthreads();
    float ssum = 0.f;
    for (int i = 0; i < LL; ++i) ssum += scores[i];
    const float sinv = 1.f / ssum;

    // xs[c] = sum_i x[c,i] * softmax(q)[i]
    {
        float s = 0.f;
        #pragma unroll 4
        for (int i = 0; i < LL; ++i)
            s += xrow[tid * XROW_STRIDE + i] * scores[i];
        xsum[tid] = s * sinv;
    }
    __syncthreads();

    // ctx[c] = Wk[c,:] . xs + bk[c]   (Wk = rows 1..C)
    {
        float cacc = qkv_b[1 + tid];
        const float* wkrow = qkv_w + (size_t)(1 + tid) * CC;
        #pragma unroll 4
        for (int cp = 0; cp < CC; ++cp)
            cacc += wkrow[cp] * xsum[cp];
        ctxv[tid] = cacc;
    }
    __syncthreads();

    // v = Wv @ x_slice : [256x256] @ [256x80], WMMA bf16 -> f32
    // 16 M-tiles x 5 N-tiles = 80 tile-pairs over 8 waves, 8 K-steps each.
    const int wave = tid >> 5;
    const int lane = tid & 31;
    const int n16  = lane & 15;
    const int kh   = lane >> 4;

    for (int p = wave * 10; p < wave * 10 + 10; ++p) {
        const int mt = p / 5, nt = p % 5;
        v8f acc = {};
        const __bf16* brow = &xT[(nt * 16 + n16) * XT_STRIDE];
        #pragma unroll
        for (int kt = 0; kt < 8; ++kt) {
            const __bf16* ap = wv_tiled + ((size_t)(mt * 8 + kt) * 32 + lane) * 16;
            v8bf a0 = *(const v8bf*)(ap);
            v8bf a1 = *(const v8bf*)(ap + 8);
            v8bf b0 = *(const v8bf*)(brow + kt * 32 + kh * 8);
            v8bf b1 = *(const v8bf*)(brow + kt * 32 + 16 + kh * 8);
            v16bf av = __builtin_shufflevector(a0, a1, 0,1,2,3,4,5,6,7,8,9,10,11,12,13,14,15);
            v16bf bv = __builtin_shufflevector(b0, b1, 0,1,2,3,4,5,6,7,8,9,10,11,12,13,14,15);
            acc = __builtin_amdgcn_wmma_f32_16x16x32_bf16(
                false, av, false, bv, (short)0, acc, false, false);
        }
        // Epilogue: out = x + sigmoid(v + bv) * ctx
        #pragma unroll
        for (int r = 0; r < 8; ++r) {
            const int m  = r + kh * 8;           // D layout: VGPR r -> M=r / r+8
            const int co = mt * 16 + m;
            const int i  = nt * 16 + n16;
            float vv  = acc[r] + qkv_b[1 + CC + co];
            float sig = 1.f / (1.f + __expf(-vv));
            float ov  = xrow[co * XROW_STRIDE + i] + sig * ctxv[co];
            dst[base + (size_t)co * PLANE + (size_t)i * istride] = ov;
        }
    }
}

// ---------------------------------------------------------------------------
// Depthwise 3x3 (SAME, zero pad) + BN(inference) + SiLU + residual.
// One block per (b,c) 80x80 plane, halo tile in LDS.
// ---------------------------------------------------------------------------
__global__ __launch_bounds__(256) void dw_bn_silu_kernel(
    const float* __restrict__ in, float* __restrict__ out,
    const float* __restrict__ dw,
    const float* __restrict__ gamma, const float* __restrict__ beta,
    const float* __restrict__ mean,  const float* __restrict__ var)
{
    __shared__ float tile[82 * 82];
    const int tid = threadIdx.x;
    const int c   = blockIdx.x % CC;
    const size_t base = (size_t)blockIdx.x * PLANE;

    for (int i = tid; i < 82 * 82; i += 256) tile[i] = 0.f;
    __syncthreads();
    for (int i = tid; i < PLANE; i += 256) {
        int h = i / WW, w = i % WW;
        tile[(h + 1) * 82 + (w + 1)] = in[base + i];
    }
    __syncthreads();

    const float k0 = dw[c*9+0], k1 = dw[c*9+1], k2 = dw[c*9+2];
    const float k3 = dw[c*9+3], k4 = dw[c*9+4], k5 = dw[c*9+5];
    const float k6 = dw[c*9+6], k7 = dw[c*9+7], k8 = dw[c*9+8];
    const float inv = rsqrtf(var[c] + BN_EPS);
    const float sc  = gamma[c] * inv;
    const float sh  = beta[c] - mean[c] * sc;

    for (int i = tid; i < PLANE; i += 256) {
        int h = i / WW, w = i % WW;
        const float* t = &tile[h * 82 + w];
        float a = k0*t[0]   + k1*t[1]   + k2*t[2]
                + k3*t[82]  + k4*t[83]  + k5*t[84]
                + k6*t[164] + k7*t[165] + k8*t[166];
        float z = a * sc + sh;
        float y = z / (1.f + __expf(-z));     // SiLU
        out[base + i] = t[83] + y;            // residual (t[83] = center)
    }
}

// ---------------------------------------------------------------------------
extern "C" void kernel_launch(void* const* d_in, const int* in_sizes, int n_in,
                              void* d_out, int out_size, void* d_ws, size_t ws_size,
                              hipStream_t stream) {
    const float* x      = (const float*)d_in[0];
    const float* qkvwW  = (const float*)d_in[1];
    const float* qkvbW  = (const float*)d_in[2];
    const float* dwW    = (const float*)d_in[3];
    const float* gammaW = (const float*)d_in[4];
    const float* betaW  = (const float*)d_in[5];
    const float* meanW  = (const float*)d_in[6];
    const float* varW   = (const float*)d_in[7];
    const float* qkvwH  = (const float*)d_in[8];
    const float* qkvbH  = (const float*)d_in[9];
    const float* dwH    = (const float*)d_in[10];
    const float* gammaH = (const float*)d_in[11];
    const float* betaH  = (const float*)d_in[12];
    const float* meanH  = (const float*)d_in[13];
    const float* varH   = (const float*)d_in[14];

    float* tmp = (float*)d_ws;
    const size_t tmpBytes = (size_t)BB * CC * PLANE * sizeof(float);  // ~105 MB
    __bf16* wvT = (__bf16*)((char*)d_ws + tmpBytes);                  // [2][C*C] bf16
    (void)ws_size; (void)in_sizes; (void)n_in; (void)out_size;

    // Repack Wv for both axes into WMMA-friendly bf16 tiles
    prep_wv_kernel<<<(2 * CC * CC + 255) / 256, 256, 0, stream>>>(qkvwW, qkvwH, wvT);

    // Pass 1: attention along W, then dwconv+BN+SiLU+residual
    attn_axis_kernel<<<BB * LL, 256, 0, stream>>>(x, tmp, qkvwW, qkvbW, wvT,
                                                  /*rowMul=*/WW, /*istride=*/1);
    dw_bn_silu_kernel<<<BB * CC, 256, 0, stream>>>(tmp, (float*)d_out,
                                                   dwW, gammaW, betaW, meanW, varW);

    // Pass 2: attention along H, then dwconv+BN+SiLU+residual
    attn_axis_kernel<<<BB * LL, 256, 0, stream>>>((const float*)d_out, tmp,
                                                  qkvwH, qkvbH, wvT + (size_t)CC * CC,
                                                  /*rowMul=*/1, /*istride=*/WW);
    dw_bn_silu_kernel<<<BB * CC, 256, 0, stream>>>(tmp, (float*)d_out,
                                                   dwH, gammaH, betaH, meanH, varH);
}
